// DeepFaceCross_55155970015759
// MI455X (gfx1250) — matrix operand order repaired and checked by reading, not compiled, not run on hardware
//
#include <hip/hip_runtime.h>
#include <hip/hip_bf16.h>
#include <hip/hip_fp16.h>

typedef __attribute__((ext_vector_type(16))) _Float16 v16h;
typedef __attribute__((ext_vector_type(8)))  _Float16 v8h;
typedef __attribute__((ext_vector_type(8)))  float    v8f;

// ---------------------------------------------------------------------------
// Network dims (batch 32)
//   conv1: [32,152,152,3] * [11,11,3,32] -> [32,142,142,32], relu
//   pool : 3x3 s2 -> [32,70,70,32]
//   conv3: [9,9,32,16] -> [32,62,62,16], relu        (implicit GEMM, WMMA)
//   LC4  : k=9 s=1 -> [32,54,54,16]   (per-pixel GEMM, WMMA, split-K x2)
//   LC5  : k=7 s=2 -> [32,24,24,16]   (split-K x4)
//   LC6  : k=5 s=1 -> [32,20,20,16]   (split-K x4)
//   FC7  : [32,6400]@[6400,4096], relu (WMMA, split-K x8)
//   FC8  : [32,4096]@[4096,1], sigmoid
// ---------------------------------------------------------------------------

// ---------------- conv1 + relu, f32 in, f16 out (direct, VALU) -------------
__global__ void conv1_relu_f16(const float* __restrict__ x,
                               const float* __restrict__ W,
                               const float* __restrict__ bias,
                               _Float16* __restrict__ y) {
  const int NPOS = 32 * 142 * 142;
  int id = blockIdx.x * blockDim.x + threadIdx.x;
  if (id >= NPOS) return;
  int ox = id % 142;
  int t  = id / 142;
  int oy = t % 142;
  int b  = t / 142;

  float acc[32];
#pragma unroll
  for (int c = 0; c < 32; ++c) acc[c] = bias[c];

  const float* xb = x + (((size_t)b * 152 + oy) * 152 + ox) * 3;
  for (int ky = 0; ky < 11; ++ky) {
    const float* xr = xb + (size_t)ky * 152 * 3;
    const float* wr = W + (size_t)ky * 11 * 3 * 32;
    for (int kx = 0; kx < 11; ++kx) {
      float i0 = xr[kx * 3 + 0];
      float i1 = xr[kx * 3 + 1];
      float i2 = xr[kx * 3 + 2];
      const float* w = wr + kx * 3 * 32;  // wave-uniform -> scalar loads
#pragma unroll
      for (int c = 0; c < 32; ++c)
        acc[c] = fmaf(i0, w[c], fmaf(i1, w[32 + c], fmaf(i2, w[64 + c], acc[c])));
    }
  }

  _Float16 out[32];
#pragma unroll
  for (int c = 0; c < 32; ++c) out[c] = (_Float16)fmaxf(acc[c], 0.0f);
  v8h* dst = (v8h*)(y + (size_t)id * 32);
  const v8h* src = (const v8h*)out;
  dst[0] = src[0]; dst[1] = src[1]; dst[2] = src[2]; dst[3] = src[3];
}

// ---------------- maxpool 3x3 stride 2, f16 -> f16 -------------------------
__global__ void maxpool_3x3s2(const _Float16* __restrict__ x,
                              _Float16* __restrict__ y) {
  const int N = 32 * 70 * 70 * 32;
  int id = blockIdx.x * blockDim.x + threadIdx.x;
  if (id >= N) return;
  int c  = id & 31;
  int t  = id >> 5;
  int ox = t % 70; t /= 70;
  int oy = t % 70;
  int b  = t / 70;
  const _Float16* xb = x + ((((size_t)b * 142 + oy * 2) * 142) + ox * 2) * 32 + c;
  float m = -INFINITY;
#pragma unroll
  for (int ky = 0; ky < 3; ++ky)
#pragma unroll
    for (int kx = 0; kx < 3; ++kx)
      m = fmaxf(m, (float)xb[(ky * 142 + kx) * 32]);
  y[id] = (_Float16)m;
}

// ---------------- conv3 as implicit GEMM (WMMA f16) ------------------------
// rows = (b,oy,ox) : 32*62*62 = 123008 = 7688 M-tiles; N = 16 couts;
// K = 81 chunks of 32 (the 32 input channels of one (ky,kx) tap).
__global__ void conv3_wmma(const _Float16* __restrict__ x,   // [32,70,70,32]
                           const float* __restrict__ W,      // [9,9,32,16]
                           const float* __restrict__ bias,   // [16]
                           _Float16* __restrict__ y) {       // [32,62,62,16]
  int wave = (blockIdx.x * blockDim.x + threadIdx.x) >> 5;
  if (wave >= 7688) return;                    // wave-uniform
  int lane  = threadIdx.x & 31;
  int hi    = lane >> 4;                       // K sub-block select
  int n     = lane & 15;                       // N = cout / M-row-in-tile
  int row   = wave * 16 + n;                   // A row for this lane
  int ox = row % 62;
  int t  = row / 62;
  int oy = t % 62;
  int b  = t / 62;
  const _Float16* xb = x + (((size_t)b * 70 + oy) * 70 + ox) * 32;
  int cbase = hi * 16;

  v8f acc = {};
  for (int kt = 0; kt < 81; ++kt) {
    int ky = kt / 9, kx = kt % 9;
    const _Float16* xp = xb + ((size_t)ky * 70 + kx) * 32;
    v8h lo  = *(const v8h*)(xp + cbase);
    v8h hi8 = *(const v8h*)(xp + 8 + cbase);
    v16h a;
#pragma unroll
    for (int i = 0; i < 8; ++i) { a[i] = lo[i]; a[8 + i] = hi8[i]; }

    const float* wp = W + ((size_t)kt * 32 + hi * 16) * 16 + n;
    v16h bb;
#pragma unroll
    for (int e = 0; e < 16; ++e) bb[e] = (_Float16)wp[(size_t)e * 16];

    acc = __builtin_amdgcn_wmma_f32_16x16x32_f16(false, a, false, bb,
                                                 (short)0, acc, false, false);
  }
  float bv = bias[n];
#pragma unroll
  for (int r = 0; r < 8; ++r) {
    size_t rr = (size_t)wave * 16 + r + hi * 8;
    y[rr * 16 + n] = (_Float16)fmaxf(acc[r] + bv, 0.0f);
  }
}

// ---------------- locally-connected + relu (WMMA f16, split-K) -------------
// One BLOCK per output pixel (oy,ox); SPLIT waves divide the K chunks and
// reduce partial accumulators through LDS. A = patches [32 x 16*k*k]
// (batch rows, patch index kappa = c*k*k + ky*k + kx, channel-major like
// conv_general_dilated_patches). B = LW[oy,ox] [16*k*k x 16]. Two M-tiles
// (batches 0-15 / 16-31) share each streamed B fragment.
template <int KS, int STRIDE, int SPLIT>
__global__ void lc_wmma(const _Float16* __restrict__ x,   // [32,IH,IW,16]
                        const float* __restrict__ LW,     // [OH,OW,16*KS*KS,16]
                        const float* __restrict__ Lb,     // [OH,OW,16]
                        _Float16* __restrict__ y,         // [32,OH,OW,16]
                        int IH, int IW, int OH, int OW) {
  const int KDIM   = 16 * KS * KS;
  const int NCHUNK = (KDIM + 31) / 32;
  __shared__ float red[SPLIT][2][8][32];

  int wid    = blockIdx.x;            // output pixel
  int wavein = threadIdx.x >> 5;      // K slice
  int lane   = threadIdx.x & 31;
  int hi     = lane >> 4;
  int n      = lane & 15;
  int oy = wid / OW, ox = wid % OW;

  const float* wbase = LW + (size_t)wid * KDIM * 16 + n;
  size_t img = (size_t)IH * IW * 16;
  const _Float16* xrow = x + (((size_t)(oy * STRIDE)) * IW + ox * STRIDE) * 16;

  v8f acc0 = {}, acc1 = {};
  for (int t = wavein; t < NCHUNK; t += SPLIT) {
    if (t + SPLIT < NCHUNK)
      __builtin_prefetch(wbase + (size_t)(t + SPLIT) * 32 * 16, 0, 0);

    // B fragment: row K = e + hi*16, col = n (zero-fill past KDIM)
    v16h bb;
#pragma unroll
    for (int e = 0; e < 16; ++e) {
      int kr = t * 32 + e + hi * 16;
      bb[e] = (kr < KDIM) ? (_Float16)wbase[(size_t)kr * 16] : (_Float16)0.0f;
    }

    // A fragments: lane row = batch n (tile0) / n+16 (tile1)
    v16h a0, a1;
#pragma unroll
    for (int e = 0; e < 16; ++e) {
      int eK = hi * 16 + e;
      int kk = t * 32 + eK;
      _Float16 v0 = (_Float16)0.0f, v1 = (_Float16)0.0f;
      if (kk < KDIM) {
        int c   = kk / (KS * KS);
        int pos = kk % (KS * KS);
        int ky  = pos / KS, kx = pos % KS;
        size_t off = ((size_t)ky * IW + kx) * 16 + c;
        v0 = xrow[(size_t)n * img + off];
        v1 = xrow[(size_t)(n + 16) * img + off];
      }
      a0[e] = v0; a1[e] = v1;
    }
    acc0 = __builtin_amdgcn_wmma_f32_16x16x32_f16(false, a0, false, bb,
                                                  (short)0, acc0, false, false);
    acc1 = __builtin_amdgcn_wmma_f32_16x16x32_f16(false, a1, false, bb,
                                                  (short)0, acc1, false, false);
  }

#pragma unroll
  for (int r = 0; r < 8; ++r) {
    red[wavein][0][r][lane] = acc0[r];
    red[wavein][1][r][lane] = acc1[r];
  }
  __syncthreads();

  size_t oimg = (size_t)OH * OW * 16;
  for (int v = threadIdx.x; v < 512; v += 32 * SPLIT) {
    int tile = v >> 8;
    int r    = (v >> 5) & 7;
    int ln   = v & 31;
    float s = 0.0f;
#pragma unroll
    for (int w = 0; w < SPLIT; ++w) s += red[w][tile][r][ln];
    int m  = tile * 16 + r + (ln >> 4) * 8;   // batch row
    int nn = ln & 15;                         // cout
    float val = fmaxf(s + Lb[(size_t)wid * 16 + nn], 0.0f);
    y[(size_t)m * oimg + (size_t)wid * 16 + nn] = (_Float16)val;
  }
}

// ---------------- FC7: [32,6400] @ [6400,4096] + relu (WMMA, split-K x8) ---
// One block (256 threads = 8 waves) per N-tile of 16 columns; waves divide
// the 200 K-chunks; partials reduced through LDS.
__global__ void fc7_wmma(const _Float16* __restrict__ x,   // [32,6400] f16
                         const float* __restrict__ W,      // [6400,4096]
                         const float* __restrict__ bias,   // [4096]
                         float* __restrict__ y) {          // [32,4096] f32
  const int SPLIT = 8;
  __shared__ float red[SPLIT][2][8][32];

  int wid    = blockIdx.x;             // n-tile 0..255
  int wavein = threadIdx.x >> 5;       // K slice
  int lane   = threadIdx.x & 31;
  int hi     = lane >> 4;
  int n      = lane & 15;
  int col    = wid * 16 + n;

  const float* wb = W + (size_t)(hi * 16) * 4096 + col;
  const _Float16* x0 = x + (size_t)n * 6400;
  const _Float16* x1 = x + (size_t)(n + 16) * 6400;

  v8f acc0 = {}, acc1 = {};
  for (int t = wavein; t < 200; t += SPLIT) {
    const float* wp = wb + (size_t)t * 32 * 4096;
    if (t + SPLIT < 200)
      __builtin_prefetch(wp + (size_t)SPLIT * 32 * 4096, 0, 0);
    v16h bb;
#pragma unroll
    for (int e = 0; e < 16; ++e) bb[e] = (_Float16)wp[(size_t)e * 4096];

    int kb = t * 32 + hi * 16;
    v8h l0 = *(const v8h*)(x0 + kb);
    v8h h0 = *(const v8h*)(x0 + kb + 8);
    v8h l1 = *(const v8h*)(x1 + kb);
    v8h h1 = *(const v8h*)(x1 + kb + 8);
    v16h a0, a1;
#pragma unroll
    for (int i = 0; i < 8; ++i) {
      a0[i] = l0[i]; a0[8 + i] = h0[i];
      a1[i] = l1[i]; a1[8 + i] = h1[i];
    }
    acc0 = __builtin_amdgcn_wmma_f32_16x16x32_f16(false, a0, false, bb,
                                                  (short)0, acc0, false, false);
    acc1 = __builtin_amdgcn_wmma_f32_16x16x32_f16(false, a1, false, bb,
                                                  (short)0, acc1, false, false);
  }

#pragma unroll
  for (int r = 0; r < 8; ++r) {
    red[wavein][0][r][lane] = acc0[r];
    red[wavein][1][r][lane] = acc1[r];
  }
  __syncthreads();

  for (int v = threadIdx.x; v < 512; v += 256) {
    int tile = v >> 8;
    int r    = (v >> 5) & 7;
    int ln   = v & 31;
    float s = 0.0f;
#pragma unroll
    for (int w = 0; w < SPLIT; ++w) s += red[w][tile][r][ln];
    int m  = tile * 16 + r + (ln >> 4) * 8;   // batch row
    int cc = wid * 16 + (ln & 15);            // output column
    y[(size_t)m * 4096 + cc] = fmaxf(s + bias[cc], 0.0f);
  }
}

// ---------------- FC8 + sigmoid: [32,4096] @ [4096,1] ----------------------
__global__ void fc8_sigmoid(const float* __restrict__ x,   // [32,4096]
                            const float* __restrict__ W,   // [4096,1]
                            const float* __restrict__ bias,// [1]
                            float* __restrict__ out) {     // [32]
  __shared__ float red[256];
  int b   = blockIdx.x;
  int tid = threadIdx.x;
  const float* xr = x + (size_t)b * 4096;
  float s = 0.0f;
  for (int i = tid; i < 4096; i += 256) s = fmaf(xr[i], W[i], s);
  red[tid] = s;
  __syncthreads();
  for (int k = 128; k > 0; k >>= 1) {
    if (tid < k) red[tid] += red[tid + k];
    __syncthreads();
  }
  if (tid == 0) {
    float z = red[0] + bias[0];
    out[b] = 1.0f / (1.0f + expf(-z));
  }
}

// ---------------------------------------------------------------------------
extern "C" void kernel_launch(void* const* d_in, const int* in_sizes, int n_in,
                              void* d_out, int out_size, void* d_ws, size_t ws_size,
                              hipStream_t stream) {
  (void)in_sizes; (void)n_in; (void)out_size; (void)ws_size;

  const float* x_in = (const float*)d_in[0];
  const float* W1  = (const float*)d_in[1];
  const float* b1  = (const float*)d_in[2];
  const float* W3  = (const float*)d_in[3];
  const float* b3  = (const float*)d_in[4];
  const float* LW4 = (const float*)d_in[5];
  const float* Lb4 = (const float*)d_in[6];
  const float* LW5 = (const float*)d_in[7];
  const float* Lb5 = (const float*)d_in[8];
  const float* LW6 = (const float*)d_in[9];
  const float* Lb6 = (const float*)d_in[10];
  const float* W7  = (const float*)d_in[11];
  const float* b7  = (const float*)d_in[12];
  const float* W8  = (const float*)d_in[13];
  const float* b8  = (const float*)d_in[14];
  float* out = (float*)d_out;

  // workspace carve-up (all 256B aligned)
  char* ws = (char*)d_ws;
  auto carve = [&](size_t bytes) -> char* {
    char* p = ws;
    ws += (bytes + 255) & ~(size_t)255;
    return p;
  };
  _Float16* x1 = (_Float16*)carve((size_t)32 * 142 * 142 * 32 * 2); // conv1 out
  _Float16* x2 = (_Float16*)carve((size_t)32 * 70 * 70 * 32 * 2);   // pool out
  _Float16* x3 = (_Float16*)carve((size_t)32 * 62 * 62 * 16 * 2);   // conv3 out
  _Float16* x4 = (_Float16*)carve((size_t)32 * 54 * 54 * 16 * 2);   // LC4 out
  _Float16* x5 = (_Float16*)carve((size_t)32 * 24 * 24 * 16 * 2);   // LC5 out
  _Float16* x6 = (_Float16*)carve((size_t)32 * 20 * 20 * 16 * 2);   // LC6 out
  float*    x7 = (float*)carve((size_t)32 * 4096 * 4);              // FC7 out

  // conv1: 32*142*142 = 645248 threads
  conv1_relu_f16<<<2521, 256, 0, stream>>>(x_in, W1, b1, x1);
  // maxpool: 32*70*70*32 = 5017600 threads
  maxpool_3x3s2<<<19600, 256, 0, stream>>>(x1, x2);
  // conv3: 7688 waves, 4 waves/block
  conv3_wmma<<<1922, 128, 0, stream>>>(x2, W3, b3, x3);
  // LC layers: one block per output pixel, SPLIT waves split the K chunks
  lc_wmma<9, 1, 2><<<54 * 54, 64,  0, stream>>>(x3, LW4, Lb4, x4, 62, 62, 54, 54);
  lc_wmma<7, 2, 4><<<24 * 24, 128, 0, stream>>>(x4, LW5, Lb5, x5, 54, 54, 24, 24);
  lc_wmma<5, 1, 4><<<20 * 20, 128, 0, stream>>>(x5, LW6, Lb6, x6, 24, 24, 20, 20);
  // FC7: one block (8 waves, split-K) per 16-column tile
  fc7_wmma<<<256, 256, 0, stream>>>(x6, W7, b7, x7);
  // FC8 + sigmoid: one block per batch row
  fc8_sigmoid<<<32, 256, 0, stream>>>(x7, W8, b8, out);
}
